// NTM_76776835384001
// MI455X (gfx1250) — matrix operand order repaired and checked by reading, not compiled, not run on hardware
//
#include <hip/hip_runtime.h>
#include <hip/hip_bf16.h>

// ---------------------------------------------------------------------------
// NTM forward scan for MI455X (gfx1250, wave32, WMMA).
// B=32, S=256, I=64, O=64, H=512, M=64, N=128.
// 2 persistent workgroups x 16 batch rows; GEMMs via v_wmma_f32_16x16x32_bf16.
// Activations kept in LDS as bf16 pre-swizzled WMMA A-fragments; weight
// B-fragments streamed from L2 with depth-1 software pipelining (bounded
// unroll: full unroll caused VGPR spills; rolled+rotated is spill-free).
// ---------------------------------------------------------------------------

typedef __attribute__((ext_vector_type(16))) __bf16 v16bf;
typedef __attribute__((ext_vector_type(8)))  float  v8f;

union ABf { v16bf v; unsigned short s[16]; };

#define PLD 272         // interface vector padded 268 -> 272 (17 N-tiles)
// LDS floats: actb(bf16,20*512us=5120f) c(8192) p(4352) wr(2048) ww(2048)
//             kbuf(1024) ebuf(1024) abuf(1024) wg(2048) bl(2048) bi(272) bo(64)
#define SMEM_FLOATS (5120 + 8192 + 4352 + 2048 + 2048 + 1024 + 1024 + 1024 + 2048 + 2048 + 272 + 64)

__device__ __forceinline__ unsigned short f2bfbits(float f) {
    unsigned u = __float_as_uint(f);
    u += 0x7FFFu + ((u >> 16) & 1u);           // round-to-nearest-even
    return (unsigned short)(u >> 16);
}
__device__ __forceinline__ float sigmf(float x) { return 1.0f / (1.0f + expf(-x)); }
__device__ __forceinline__ float splusf(float x) { return (x > 20.0f) ? x : log1pf(expf(x)); }

__device__ __forceinline__ float wave_sum(float v) {
#pragma unroll
    for (int o = 16; o > 0; o >>= 1) v += __shfl_xor(v, o, 32);
    return v;
}
__device__ __forceinline__ float wave_max(float v) {
#pragma unroll
    for (int o = 16; o > 0; o >>= 1) v = fmaxf(v, __shfl_xor(v, o, 32));
    return v;
}

// B-fragment load: weights pre-swizzled so each lane's 16 bf16 are contiguous.
__device__ __forceinline__ v16bf ldB(const unsigned short* base, int KT, int nt, int kt, int lane) {
    const v16bf* p = (const v16bf*)base;
    return p[(size_t)(nt * KT + kt) * 32 + lane];
}
// A-fragment load from LDS bf16 activation fragments.
__device__ __forceinline__ v16bf ldA(const unsigned short* actb, int frag, int lane) {
    const v16bf* p = (const v16bf*)actb;
    return p[frag * 32 + lane];
}

// ushort offset of activation element (row, k) inside the A-fragment layout.
// ISA 16-bit A 16x32: lane = row + 16*hi, hi = (kk>>3)&1, i = (kk&7) + ((kk&16)>>1)
__device__ __forceinline__ int actb_off(int row, int k) {
    int kt = k >> 5, kk = k & 31;
    int hi = (kk >> 3) & 1;
    int i  = (kk & 7) + ((kk & 16) >> 1);
    return kt * 512 + (row + 16 * hi) * 16 + i;
}

__device__ __forceinline__ v8f wmma_bf16(v16bf a, v16bf b, v8f c) {
    return __builtin_amdgcn_wmma_f32_16x16x32_bf16(false, a, false, b, (short)0, c, false, false);
}

// ---------------------------------------------------------------------------
// Weight pack: fp32 row-major [K][Ncols] -> bf16 WMMA B-fragments.
// frag = nt*KT + kt ; element (lane,i) = W[kt*32 + (i&7) + 8*(lane>>4) + 2*(i&8)][nt*16 + (lane&15)]
// ---------------------------------------------------------------------------
extern "C" __global__ void ntm_pack_bf16(const float* __restrict__ W,
                                         unsigned short* __restrict__ dst,
                                         int KT, int NT, int Ncols) {
    int idx = blockIdx.x * blockDim.x + threadIdx.x;
    int total = KT * NT * 512;
    if (idx >= total) return;
    int i    = idx & 15;
    int lane = (idx >> 4) & 31;
    int frag = idx >> 9;
    int kt = frag % KT;
    int nt = frag / KT;
    int hi = lane >> 4, col = lane & 15;
    int k = kt * 32 + (i & 7) + 8 * hi + ((i & 8) << 1);
    int n = nt * 16 + col;
    float v = (n < Ncols) ? W[(size_t)k * Ncols + n] : 0.0f;
    dst[idx] = f2bfbits(v);
}

// ---------------------------------------------------------------------------
// NTM addressing head (one wave == one batch row).
// ---------------------------------------------------------------------------
__device__ void ntm_address(int lane, const float* pp, float* wstate,
                            const float* memb, float* kbuf, float* wgbuf,
                            float* wloc) {
    float beta  = splusf(pp[64]);
    float g     = sigmf(pp[65]);
    float s0r = pp[66], s1r = pp[67], s2r = pp[68];
    float smx = fmaxf(s0r, fmaxf(s1r, s2r));
    float e0 = expf(s0r - smx), e1 = expf(s1r - smx), e2 = expf(s2r - smx);
    float einv = 1.0f / (e0 + e1 + e2);
    float s0 = e0 * einv, s1 = e1 * einv, s2 = e2 * einv;
    float gamma = 1.0f + splusf(pp[69]);

    kbuf[lane]      = tanhf(pp[lane]);
    kbuf[lane + 32] = tanhf(pp[lane + 32]);
    __syncthreads();

    float kn = kbuf[lane] * kbuf[lane] + kbuf[lane + 32] * kbuf[lane + 32];
    float knorm = sqrtf(wave_sum(kn));

    float sim[4];
#pragma unroll
    for (int t = 0; t < 4; ++t) {
        int n = t * 32 + lane;
        const float* mrow = memb + n * 64;
        float dot = 0.0f, mn = 0.0f;
#pragma unroll 4
        for (int m = 0; m < 64; ++m) {
            float mv = mrow[m];
            dot += mv * kbuf[m];
            mn  += mv * mv;
        }
        sim[t] = dot / (knorm * sqrtf(mn) + 1e-8f);
    }
    // softmax(beta*sim) over N=128
    float mx = -1e30f;
#pragma unroll
    for (int t = 0; t < 4; ++t) { sim[t] *= beta; mx = fmaxf(mx, sim[t]); }
    mx = wave_max(mx);
    float sum = 0.0f;
#pragma unroll
    for (int t = 0; t < 4; ++t) { sim[t] = expf(sim[t] - mx); sum += sim[t]; }
    sum = wave_sum(sum);
    float inv = 1.0f / sum;
#pragma unroll
    for (int t = 0; t < 4; ++t) {
        int n = t * 32 + lane;
        float wgv = g * (sim[t] * inv) + (1.0f - g) * wstate[n];
        wgbuf[n] = wgv;
    }
    __syncthreads();
    // circular shift {-1,0,+1} then sharpen
    float wp[4]; float psum = 0.0f;
#pragma unroll
    for (int t = 0; t < 4; ++t) {
        int n = t * 32 + lane;
        float wsv = s0 * wgbuf[(n + 1) & 127] + s1 * wgbuf[n] + s2 * wgbuf[(n - 1) & 127];
        float v = powf(fmaxf(wsv, 1e-12f), gamma);
        wp[t] = v; psum += v;
    }
    psum = wave_sum(psum);
    float rinv = 1.0f / psum;
#pragma unroll
    for (int t = 0; t < 4; ++t) {
        int n = t * 32 + lane;
        float v = wp[t] * rinv;
        wstate[n] = v;
        wloc[t] = v;
    }
}

// ---------------------------------------------------------------------------
// Main persistent kernel: 2 blocks x 512 threads (16 wave32). Block handles
// batch rows [blk*16, blk*16+16) across all S=256 steps.
// ---------------------------------------------------------------------------
extern "C" __global__ void __launch_bounds__(512, 1)
ntm_main(const float* __restrict__ inputs,
         const float* __restrict__ b_lstm,
         const float* __restrict__ b_int,
         const float* __restrict__ b_out,
         const unsigned short* __restrict__ Wl,   // packed bf16 [128 nt][20 kt]
         const unsigned short* __restrict__ Wi,   // packed bf16 [17 nt][16 kt]
         const unsigned short* __restrict__ Wo,   // packed bf16 [4 nt][18 kt]
         float* __restrict__ memg,                // [32][128][64] scratch
         float* __restrict__ out)                 // [32][256][64]
{
    extern __shared__ float sm[];
    unsigned short* actb = (unsigned short*)sm;   // 20 frags x 512 us (bf16 A-frags)
    float* c_s    = sm     + 5120;                // 16 x 512
    float* p_s    = c_s    + 8192;                // 16 x PLD
    float* wr_s   = p_s    + 16 * PLD;            // 16 x 128
    float* ww_s   = wr_s   + 2048;                // 16 x 128
    float* kbuf_s = ww_s   + 2048;                // 16 x 64
    float* ebuf_s = kbuf_s + 1024;                // 16 x 64
    float* abuf_s = ebuf_s + 1024;                // 16 x 64
    float* wg_s   = abuf_s + 1024;                // 16 x 128
    float* bl_s   = wg_s   + 2048;                // 2048
    float* bi_s   = bl_s   + 2048;                // 272
    float* bo_s   = bi_s   + 272;                 // 64

    const int blk  = blockIdx.x;
    const int tid  = threadIdx.x;
    const int w    = tid >> 5;       // wave id 0..15  (== local batch row)
    const int lane = tid & 31;
    const int hi   = lane >> 4;
    const int colL = lane & 15;
    const v8f vzero = {0.f, 0.f, 0.f, 0.f, 0.f, 0.f, 0.f, 0.f};

    // ----- initial state -----
    for (int e = tid; e < 20 * 512; e += 512) actb[e] = 0;    // x,r,h = 0
    for (int e = tid; e < 16 * 512; e += 512) c_s[e] = 0.0f;
    for (int e = tid; e < 16 * 128; e += 512) {
        float v = ((e & 127) == 0) ? 1.0f : 0.0f;             // one-hot w0
        wr_s[e] = v; ww_s[e] = v;
    }
    for (int e = tid; e < 2048; e += 512) bl_s[e] = b_lstm[e];
    if (tid < 272) bi_s[tid] = (tid < 268) ? b_int[tid] : 0.0f;
    if (tid < 64)  bo_s[tid] = b_out[tid];
    float* memblk = memg + (size_t)blk * 16 * 8192;
    for (int e = tid; e < 16 * 8192; e += 512) memblk[e] = 0.01f;
    __syncthreads();
    {   // r0 = 0.01 (after zero-fill; disjoint partitioning needs the barrier)
        unsigned short rb = f2bfbits(0.01f);
        for (int e = tid; e < 16 * 64; e += 512)
            actb[actb_off(e >> 6, 64 + (e & 63))] = rb;
    }
    __syncthreads();

    for (int st = 0; st < 256; ++st) {
        // ---- load x_t into act k=[0,64) as bf16 fragments ----
        for (int e = tid; e < 16 * 64; e += 512) {
            int row = e >> 6, ii = e & 63;
            actb[actb_off(row, ii)] =
                f2bfbits(inputs[((size_t)(blk * 16 + row) * 256 + st) * 64 + ii]);
        }
        __syncthreads();

        // ---- LSTM GEMM: z = act[16x640] @ Wl[640x2048], fused gates ----
        // wave w owns z column tiles {g*32 + 2w + j} for g in {i,f,g,o}, j in {0,1}
        // depth-1 software pipeline: prefetch A/B of kt+1 before WMMAs of kt
        v8f acc[2][4];
#pragma unroll
        for (int j = 0; j < 2; ++j) {
            int tbase = 2 * w + j;
            v8f a0 = vzero, a1 = vzero, a2 = vzero, a3 = vzero;
            v16bf aCur = ldA(actb, 0, lane);
            ABf b0, b1, b2, b3;
            b0.v = ldB(Wl, 20, tbase,      0, lane);
            b1.v = ldB(Wl, 20, 32 + tbase, 0, lane);
            b2.v = ldB(Wl, 20, 64 + tbase, 0, lane);
            b3.v = ldB(Wl, 20, 96 + tbase, 0, lane);
#pragma unroll 2
            for (int kt = 0; kt < 20; ++kt) {
                v16bf aNext = aCur;
                ABf n0 = b0, n1 = b1, n2 = b2, n3 = b3;
                if (kt < 19) {
                    aNext = ldA(actb, kt + 1, lane);
                    n0.v = ldB(Wl, 20, tbase,      kt + 1, lane);
                    n1.v = ldB(Wl, 20, 32 + tbase, kt + 1, lane);
                    n2.v = ldB(Wl, 20, 64 + tbase, kt + 1, lane);
                    n3.v = ldB(Wl, 20, 96 + tbase, kt + 1, lane);
                }
                a0 = wmma_bf16(aCur, b0.v, a0);
                a1 = wmma_bf16(aCur, b1.v, a1);
                a2 = wmma_bf16(aCur, b2.v, a2);
                a3 = wmma_bf16(aCur, b3.v, a3);
                aCur = aNext; b0 = n0; b1 = n1; b2 = n2; b3 = n3;
            }
            acc[j][0] = a0; acc[j][1] = a1; acc[j][2] = a2; acc[j][3] = a3;
        }
        __syncthreads();   // all waves done reading actb before h is rewritten
#pragma unroll
        for (int j = 0; j < 2; ++j) {
            int hcol = (2 * w + j) * 16 + colL;
            float bi  = bl_s[hcol];
            float bfv = bl_s[512 + hcol];
            float bgv = bl_s[1024 + hcol];
            float bov = bl_s[1536 + hcol];
#pragma unroll
            for (int v = 0; v < 8; ++v) {
                int row = v + 8 * hi;
                float zi = acc[j][0][v] + bi;
                float zf = acc[j][1][v] + bfv;
                float zg = acc[j][2][v] + bgv;
                float zo = acc[j][3][v] + bov;
                float cp = c_s[row * 512 + hcol];
                float cn = sigmf(zf) * cp + sigmf(zi) * tanhf(zg);
                float hn = sigmf(zo) * tanhf(cn);
                c_s[row * 512 + hcol] = cn;
                actb[actb_off(row, 128 + hcol)] = f2bfbits(hn);
            }
        }
        __syncthreads();

        // ---- interface GEMM: p = h[16x512] @ Wi[512x272]; h is frags 4..19 ----
        {
            int ntiles = (w == 0) ? 2 : 1;
            for (int pt = 0; pt < ntiles; ++pt) {
                int nt = (pt == 0) ? w : 16;
                v8f pa = vzero;
                v16bf aCur = ldA(actb, 4, lane);
                ABf bc; bc.v = ldB(Wi, 16, nt, 0, lane);
#pragma unroll 2
                for (int kt = 0; kt < 16; ++kt) {
                    v16bf aNext = aCur;
                    ABf bn = bc;
                    if (kt < 15) {
                        aNext = ldA(actb, 5 + kt, lane);
                        bn.v  = ldB(Wi, 16, nt, kt + 1, lane);
                    }
                    pa = wmma_bf16(aCur, bc.v, pa);
                    aCur = aNext; bc = bn;
                }
                int n = nt * 16 + colL;
                float bb = bi_s[n];
#pragma unroll
                for (int v = 0; v < 8; ++v) {
                    int row = v + 8 * hi;
                    p_s[row * PLD + n] = pa[v] + bb;
                }
            }
        }
        __syncthreads();

        // ---- addressing + memory update + read (wave w == batch row w) ----
        {
            const int bb = w;
            float* pp    = p_s + bb * PLD;
            float* memb  = memg + (size_t)(blk * 16 + bb) * 8192;
            float* kb    = kbuf_s + bb * 64;
            float* wgb   = wg_s + bb * 128;
            float wloc[4];

            // write head (params at offset R_SZ=70)
            ntm_address(lane, pp + 70, ww_s + bb * 128, memb, kb, wgb, wloc);

            // erase / add vectors
            float* eb = ebuf_s + bb * 64;
            float* ab = abuf_s + bb * 64;
            eb[lane]      = sigmf(pp[140 + lane]);
            eb[lane + 32] = sigmf(pp[140 + lane + 32]);
            ab[lane]      = tanhf(pp[204 + lane]);
            ab[lane + 32] = tanhf(pp[204 + lane + 32]);
            __syncthreads();

#pragma unroll
            for (int t = 0; t < 4; ++t) {
                int n = t * 32 + lane;
                float wn = wloc[t];
                float* mrow = memb + n * 64;
#pragma unroll 4
                for (int m = 0; m < 64; ++m)
                    mrow[m] = mrow[m] * (1.0f - wn * eb[m]) + wn * ab[m];
            }

            // read head on updated memory
            ntm_address(lane, pp, wr_s + bb * 128, memb, kb, wgb, wloc);

            // r = w_r @ mem  -> act k=[64,128) (frags 2..3)
            float* wrp = wr_s + bb * 128;
#pragma unroll
            for (int rep = 0; rep < 2; ++rep) {
                int m = lane + 32 * rep;
                float racc = 0.0f;
#pragma unroll 4
                for (int n = 0; n < 128; ++n) racc += wrp[n] * memb[n * 64 + m];
                actb[actb_off(bb, 64 + m)] = f2bfbits(racc);
            }
        }
        __syncthreads();

        // ---- output GEMM: [h|r][16x576] @ Wo[576x64] (waves 0..3) ----
        // out k-tile kt -> act frag 4+kt (h, kt<16) or 2+(kt-16) (r)
        if (w < 4) {
            v8f oa = vzero;
            v16bf aCur = ldA(actb, 4, lane);
            ABf bc; bc.v = ldB(Wo, 18, w, 0, lane);
#pragma unroll 2
            for (int kt = 0; kt < 18; ++kt) {
                v16bf aNext = aCur;
                ABf bn = bc;
                if (kt < 17) {
                    int nf = (kt + 1 < 16) ? (5 + kt) : (2 + (kt + 1 - 16));
                    aNext = ldA(actb, nf, lane);
                    bn.v  = ldB(Wo, 18, w, kt + 1, lane);
                }
                oa = wmma_bf16(aCur, bc.v, oa);
                aCur = aNext; bc = bn;
            }
            int n = w * 16 + colL;
            float bo = bo_s[n];
#pragma unroll
            for (int v = 0; v < 8; ++v) {
                int row = v + 8 * hi;
                out[((size_t)(blk * 16 + row) * 256 + st) * 64 + n] = oa[v] + bo;
            }
        }
        __syncthreads();
    }
}

// ---------------------------------------------------------------------------
// Host launcher
// ---------------------------------------------------------------------------
extern "C" void kernel_launch(void* const* d_in, const int* in_sizes, int n_in,
                              void* d_out, int out_size, void* d_ws, size_t ws_size,
                              hipStream_t stream) {
    (void)in_sizes; (void)n_in; (void)out_size; (void)ws_size;
    const float* inputs = (const float*)d_in[0];
    const float* W_lstm = (const float*)d_in[1];
    const float* b_lstm = (const float*)d_in[2];
    const float* W_int  = (const float*)d_in[3];
    const float* b_int  = (const float*)d_in[4];
    const float* W_out  = (const float*)d_in[5];
    const float* b_out  = (const float*)d_in[6];

    // workspace layout (256B-aligned sections)
    unsigned short* wl = (unsigned short*)d_ws;                           // 2,621,440 B
    unsigned short* wi = (unsigned short*)((char*)d_ws + 2621440);        //   278,528 B
    unsigned short* wo = (unsigned short*)((char*)d_ws + 2899968);        //    73,728 B
    float*        memg = (float*)((char*)d_ws + 2973696);                 // 1,048,576 B

    // pack weights to bf16 WMMA fragments
    {
        int tot = 20 * 128 * 512;
        ntm_pack_bf16<<<(tot + 255) / 256, 256, 0, stream>>>(W_lstm, wl, 20, 128, 2048);
    }
    {
        int tot = 16 * 17 * 512;
        ntm_pack_bf16<<<(tot + 255) / 256, 256, 0, stream>>>(W_int, wi, 16, 17, 268);
    }
    {
        int tot = 18 * 4 * 512;
        ntm_pack_bf16<<<(tot + 255) / 256, 256, 0, stream>>>(W_out, wo, 18, 4, 64);
    }

    size_t smem = (size_t)SMEM_FLOATS * sizeof(float);   // 117,056 B < 320 KB/WGP
    hipFuncSetAttribute((const void*)ntm_main,
                        hipFuncAttributeMaxDynamicSharedMemorySize, (int)smem);
    ntm_main<<<dim3(2), dim3(512), smem, stream>>>(
        inputs, b_lstm, b_int, b_out, wl, wi, wo, memg, (float*)d_out);
}
